// PointCloudEncoder_59313498357931
// MI455X (gfx1250) — matrix operand order repaired and checked by reading, not compiled, not run on hardware
//
#include <hip/hip_runtime.h>
#include <hip/hip_bf16.h>
#include <stdint.h>

// ---------------------------------------------------------------------------
// DGCNN point-cloud encoder for MI455X (gfx1250, wave32, WMMA + async-to-LDS).
// Heavy parts (f@f^T Gram for kNN, edge-feature GEMM) run on
// v_wmma_f32_16x16x32_f16 with column tiles / weights staged into LDS via
// GLOBAL_LOAD_ASYNC_TO_LDS (double-buffered in the kNN hot loop).
// ---------------------------------------------------------------------------

#define KNN 10

typedef __attribute__((ext_vector_type(16))) _Float16 v16h;
typedef __attribute__((ext_vector_type(8)))  float    v8f;

// Low 32 bits of a generic pointer to __shared__ = AS3 byte offset (what the
// async-to-LDS instruction's VDST operand wants).
__device__ __forceinline__ unsigned lds_off(const void* p) {
  return (unsigned)(uintptr_t)p;
}

// ---------------------------------------------------------------- top-k util
__device__ __forceinline__ void topk_insert(float d, int j,
                                            float (&best)[KNN], int (&bid)[KNN]) {
  if (d < best[KNN - 1]) {
    best[KNN - 1] = d; bid[KNN - 1] = j;
    #pragma unroll
    for (int q = KNN - 1; q > 0; --q) {     // one bubble pass keeps it sorted
      if (best[q] < best[q - 1]) {
        float tf = best[q]; best[q] = best[q - 1]; best[q - 1] = tf;
        int   ti = bid[q];  bid[q]  = bid[q - 1];  bid[q - 1]  = ti;
      }
    }
  }
}

// ------------------------------------------------------- layer-1 kNN (3-dim)
__global__ void __launch_bounds__(256)
knn3_kernel(const float* __restrict__ x, int* __restrict__ idx, int N) {
  __shared__ float sx[256], sy[256], sz[256];
  const int row = blockIdx.x * 256 + threadIdx.x;
  const float px = x[row * 3 + 0], py = x[row * 3 + 1], pz = x[row * 3 + 2];
  float best[KNN]; int bid[KNN];
  #pragma unroll
  for (int i = 0; i < KNN; ++i) { best[i] = 3.4e38f; bid[i] = 0; }

  for (int t = 0; t < N; t += 256) {
    const int j = t + threadIdx.x;
    sx[threadIdx.x] = x[j * 3 + 0];
    sy[threadIdx.x] = x[j * 3 + 1];
    sz[threadIdx.x] = x[j * 3 + 2];
    __syncthreads();
    for (int c = 0; c < 256; ++c) {
      const float dx = px - sx[c], dy = py - sy[c], dz = pz - sz[c];
      topk_insert(dx * dx + dy * dy + dz * dz, t + c, best, bid);
    }
    __syncthreads();
  }
  #pragma unroll
  for (int k = 0; k < KNN; ++k) idx[row * KNN + k] = bid[k];
}

// --------------------------------------------- 64-dim kNN via WMMA Gram tiles
// Block: 128 threads (4 waves) owning 128 rows.  Column loop in tiles of 32.
// The 32x64 f16 column tile is async-DMA'd into LDS (double-buffered; the
// copy of tile t+1 overlaps the WMMA + top-k scan of tile t).  Each wave
// computes two 16x16 Gram tiles per column sub-tile with chained
// v_wmma_f32_16x16x32_f16 (K=64), converts to squared distance, stores to LDS,
// then every thread scans its own row and maintains a sorted top-10.
__global__ void __launch_bounds__(128)
knn64_kernel(const _Float16* __restrict__ fh, const float* __restrict__ sq,
             int* __restrict__ idx, int N) {
  __shared__ float dtile[128][33];
  __shared__ __align__(16) _Float16 ctile[2][32 * 64];   // 2 x 4KB column tiles

  const int lane = threadIdx.x & 31;
  const int w    = threadIdx.x >> 5;
  const int m    = lane & 15;
  const int hi   = lane >> 4;
  const int r0   = blockIdx.x * 128;

  // A fragments for this wave's 32 rows (two 16-row tiles), K=0..63.
  v16h  afrag[2][2];
  float sqr[2][8];
  #pragma unroll
  for (int rt = 0; rt < 2; ++rt) {
    const int row = r0 + w * 32 + rt * 16 + m;
    const _Float16* fr = fh + (size_t)row * 64;
    #pragma unroll
    for (int ks = 0; ks < 2; ++ks) {
      const int kb = ks * 32;
      uint4 q0 = *(const uint4*)(fr + kb + hi * 8);
      uint4 q1 = *(const uint4*)(fr + kb + 16 + hi * 8);
      v16h a; ((uint4*)&a)[0] = q0; ((uint4*)&a)[1] = q1;
      afrag[rt][ks] = a;
    }
    #pragma unroll
    for (int v = 0; v < 8; ++v)
      sqr[rt][v] = sq[r0 + w * 32 + rt * 16 + hi * 8 + v];
  }

  float best[KNN]; int bid[KNN];
  #pragma unroll
  for (int i = 0; i < KNN; ++i) { best[i] = 3.4e38f; bid[i] = 0; }

  // Async copy of one 4KB column tile: 128 threads x 32B (two b128 each).
  auto issue_copy = [&](int c0, int buf) {
    const unsigned lb = lds_off(&ctile[buf][0]) + (unsigned)(threadIdx.x * 32u);
    const unsigned long long ga = (unsigned long long)(uintptr_t)fh +
                                  (unsigned long long)c0 * 128ull +
                                  (unsigned long long)(threadIdx.x * 32u);
    asm volatile(
        "global_load_async_to_lds_b128 %0, %1, off\n\t"
        "global_load_async_to_lds_b128 %0, %1, off offset:16"
        :: "v"(lb), "v"(ga)
        : "memory");
  };

  issue_copy(0, 0);
  for (int c0 = 0; c0 < N; c0 += 32) {
    const int buf = (c0 >> 5) & 1;
    if (c0 + 32 < N) {
      issue_copy(c0 + 32, buf ^ 1);
      asm volatile("s_wait_asynccnt 0x2" ::: "memory");  // tile c0 landed
    } else {
      asm volatile("s_wait_asynccnt 0x0" ::: "memory");  // last tile landed
    }
    __syncthreads();                                     // visible to all waves

    #pragma unroll
    for (int csi = 0; csi < 2; ++csi) {
      const int cs  = c0 + csi * 16;
      const float sqc = sq[cs + m];
      // B column n is feature row (cs+n): read it from the staged LDS tile.
      const _Float16* fc = &ctile[buf][(csi * 16 + m) * 64 + hi * 8];
      v16h bfrag[2];
      #pragma unroll
      for (int ks = 0; ks < 2; ++ks) {
        uint4 q0 = *(const uint4*)(fc + ks * 32);
        uint4 q1 = *(const uint4*)(fc + ks * 32 + 16);
        v16h b; ((uint4*)&b)[0] = q0; ((uint4*)&b)[1] = q1;
        bfrag[ks] = b;
      }
      #pragma unroll
      for (int rt = 0; rt < 2; ++rt) {
        v8f acc = {0.f, 0.f, 0.f, 0.f, 0.f, 0.f, 0.f, 0.f};
        acc = __builtin_amdgcn_wmma_f32_16x16x32_f16(false, afrag[rt][0], false,
                                                     bfrag[0], (short)0, acc,
                                                     false, false);
        acc = __builtin_amdgcn_wmma_f32_16x16x32_f16(false, afrag[rt][1], false,
                                                     bfrag[1], (short)0, acc,
                                                     false, false);
        const int rl = w * 32 + rt * 16 + hi * 8;
        const int cl = csi * 16 + m;
        #pragma unroll
        for (int v = 0; v < 8; ++v)
          dtile[rl + v][cl] = sqr[rt][v] + sqc - 2.0f * acc[v];
      }
    }
    __syncthreads();
    const int t = threadIdx.x;
    #pragma unroll 4
    for (int c = 0; c < 32; ++c)
      topk_insert(dtile[t][c], c0 + c, best, bid);
    __syncthreads();
  }
  const int row = r0 + threadIdx.x;
  #pragma unroll
  for (int k = 0; k < KNN; ++k) idx[row * KNN + k] = bid[k];
}

// ------------------------------------------- layer-1 edge MLP (2C=6, no WMMA)
__global__ void __launch_bounds__(256)
edge6_kernel(const float* __restrict__ x, const int* __restrict__ idx,
             const float* __restrict__ W, const float* __restrict__ b,
             float* __restrict__ hbuf, int NK) {
  __shared__ float sW[6 * 64];
  __shared__ float sb[64];
  for (int i = threadIdx.x; i < 6 * 64; i += 256) sW[i] = W[i];
  if (threadIdx.x < 64) sb[threadIdx.x] = b[threadIdx.x];
  __syncthreads();

  const int r = blockIdx.x * 256 + threadIdx.x;
  if (r >= NK) return;
  const int n = r / KNN;
  const int j = idx[r];
  float e[6];
  e[0] = x[n * 3 + 0]; e[1] = x[n * 3 + 1]; e[2] = x[n * 3 + 2];
  e[3] = x[j * 3 + 0] - e[0];
  e[4] = x[j * 3 + 1] - e[1];
  e[5] = x[j * 3 + 2] - e[2];
  for (int c = 0; c < 64; ++c) {
    float h = sb[c];
    #pragma unroll
    for (int t = 0; t < 6; ++t) h += e[t] * sW[t * 64 + c];
    hbuf[(size_t)r * 64 + c] = h;
  }
}

// ----------------------------- edge GEMM (layers 2-4): [N*K,128] @ [128,64]
// Block: 128 threads (4 waves), wave owns one 16-row tile of edge features.
// The 16KB f16 weight matrix is async-DMA'd into LDS once per block.
__global__ void __launch_bounds__(128)
edge128_kernel(const _Float16* __restrict__ fh, const int* __restrict__ idx,
               const _Float16* __restrict__ Wh, const float* __restrict__ bias,
               float* __restrict__ hbuf) {
  __shared__ __align__(16) _Float16 sW[128 * 64];        // 16KB

  // Stage weights: 128 threads x 128B (eight b128 each, 2KB apart).
  {
    const unsigned lb = lds_off(&sW[0]) + (unsigned)(threadIdx.x * 16u);
    const unsigned long long ga = (unsigned long long)(uintptr_t)Wh +
                                  (unsigned long long)(threadIdx.x * 16u);
    asm volatile(
        "global_load_async_to_lds_b128 %0, %1, off\n\t"
        "global_load_async_to_lds_b128 %0, %1, off offset:2048\n\t"
        "global_load_async_to_lds_b128 %0, %1, off offset:4096\n\t"
        "global_load_async_to_lds_b128 %0, %1, off offset:6144\n\t"
        "global_load_async_to_lds_b128 %0, %1, off offset:8192\n\t"
        "global_load_async_to_lds_b128 %0, %1, off offset:10240\n\t"
        "global_load_async_to_lds_b128 %0, %1, off offset:12288\n\t"
        "global_load_async_to_lds_b128 %0, %1, off offset:14336"
        :: "v"(lb), "v"(ga)
        : "memory");
    asm volatile("s_wait_asynccnt 0x0" ::: "memory");
  }
  __syncthreads();

  const int lane  = threadIdx.x & 31;
  const int w     = threadIdx.x >> 5;
  const int m     = lane & 15;
  const int hi    = lane >> 4;
  const int rbase = (blockIdx.x * 4 + w) * 16;
  const int r     = rbase + m;
  const int n     = r / KNN;
  const int j     = idx[r];
  const _Float16* fi = fh + (size_t)n * 64;
  const _Float16* fj = fh + (size_t)j * 64;

  // Build A fragments: edge feature row = [x_i (64) | x_j - x_i (64)].
  // Every 8-half chunk stays inside one half (boundary at c=64), no divergence.
  v16h A[4];
  #pragma unroll
  for (int ks = 0; ks < 4; ++ks) {
    const int kb = ks * 32;
    v16h a;
    #pragma unroll
    for (int part = 0; part < 2; ++part) {
      const int start = kb + hi * 16 + part * 8;
      if (start < 64) {
        #pragma unroll
        for (int t = 0; t < 8; ++t) a[part * 8 + t] = fi[start + t];
      } else {
        const int c = start - 64;
        #pragma unroll
        for (int t = 0; t < 8; ++t)
          a[part * 8 + t] = (_Float16)(fj[c + t] - fi[c + t]);
      }
    }
    A[ks] = a;
  }

  #pragma unroll
  for (int csi = 0; csi < 4; ++csi) {
    const int col = csi * 16 + m;
    v8f acc = {0.f, 0.f, 0.f, 0.f, 0.f, 0.f, 0.f, 0.f};
    #pragma unroll
    for (int ks = 0; ks < 4; ++ks) {
      const int kb = ks * 32;
      v16h b;
      #pragma unroll
      for (int e = 0; e < 16; ++e)
        b[e] = sW[(kb + hi * 16 + e) * 64 + col];
      acc = __builtin_amdgcn_wmma_f32_16x16x32_f16(false, A[ks], false, b,
                                                   (short)0, acc, false, false);
    }
    const float bb = bias[col];
    #pragma unroll
    for (int v = 0; v < 8; ++v)
      hbuf[(size_t)(rbase + hi * 8 + v) * 64 + col] = acc[v] + bb;
  }
}

// -------------------------------------------- BN batch stats (block/channel)
__global__ void __launch_bounds__(256)
stats_kernel(const float* __restrict__ hbuf, int NK,
             float* __restrict__ sums, float* __restrict__ sumsq) {
  const int c = blockIdx.x;            // 0..63
  float s = 0.f, s2 = 0.f;
  for (int i = threadIdx.x; i < NK; i += 256) {
    const float v = hbuf[(size_t)i * 64 + c];
    s += v; s2 += v * v;
  }
  __shared__ float rs[256], rq[256];
  rs[threadIdx.x] = s; rq[threadIdx.x] = s2;
  __syncthreads();
  for (int off = 128; off > 0; off >>= 1) {
    if (threadIdx.x < off) {
      rs[threadIdx.x] += rs[threadIdx.x + off];
      rq[threadIdx.x] += rq[threadIdx.x + off];
    }
    __syncthreads();
  }
  if (threadIdx.x == 0) { sums[c] = rs[0]; sumsq[c] = rq[0]; }
}

__global__ void bnfin_kernel(const float* __restrict__ sums,
                             const float* __restrict__ sumsq,
                             const float* __restrict__ g,
                             const float* __restrict__ be, float invNK,
                             float* __restrict__ scale,
                             float* __restrict__ shift) {
  const int c = threadIdx.x;           // 64 threads
  const float mu  = sums[c] * invNK;
  const float var = sumsq[c] * invNK - mu * mu;
  const float sc  = g[c] * rsqrtf(var + 1e-5f);
  scale[c] = sc;
  shift[c] = be[c] - mu * sc;
}

// ---------------------------------------- (BN) -> ReLU -> max over K per point
__global__ void __launch_bounds__(64)
relumax_kernel(const float* __restrict__ hbuf, const float* __restrict__ scale,
               const float* __restrict__ shift, int use_bn,
               float* __restrict__ f) {
  const int n = blockIdx.x;
  const int c = threadIdx.x;           // 64 threads
  const float sc = use_bn ? scale[c] : 1.f;
  const float sh = use_bn ? shift[c] : 0.f;
  float mx = -3.4e38f;
  #pragma unroll
  for (int k = 0; k < KNN; ++k) {
    float v = hbuf[((size_t)n * KNN + k) * 64 + c];
    v = fmaxf(v * sc + sh, 0.f);
    mx = fmaxf(mx, v);
  }
  f[(size_t)n * 64 + c] = mx;
}

// -------------------------------------- f32 -> f16 copy + squared norms
__global__ void __launch_bounds__(256)
castnorm_kernel(const float* __restrict__ f, _Float16* __restrict__ fh,
                float* __restrict__ sq, int N) {
  const int n = blockIdx.x * 256 + threadIdx.x;
  if (n >= N) return;
  float s = 0.f;
  #pragma unroll 8
  for (int c = 0; c < 64; ++c) {
    const float v = f[(size_t)n * 64 + c];
    s += v * v;
    fh[(size_t)n * 64 + c] = (_Float16)v;
  }
  sq[n] = s;
}

__global__ void __launch_bounds__(256)
castw_kernel(const float* __restrict__ W, _Float16* __restrict__ Wh, int n) {
  const int i = blockIdx.x * 256 + threadIdx.x;
  if (i < n) Wh[i] = (_Float16)W[i];
}

// ------------------------------------------------------- segment max + concat
__global__ void __launch_bounds__(256)
segmax_kernel(const float* __restrict__ f1, const float* __restrict__ f2,
              const float* __restrict__ f3, const float* __restrict__ f4,
              float* __restrict__ p, int pts) {
  const int b = blockIdx.x;            // B segments
  const int c = threadIdx.x;           // 256 output channels
  const float* f = (c < 64) ? f1 : (c < 128) ? f2 : (c < 192) ? f3 : f4;
  const int ch = c & 63;
  const int base = b * pts;
  float mx = -3.4e38f;
  for (int i = 0; i < pts; ++i)
    mx = fmaxf(mx, f[(size_t)(base + i) * 64 + ch]);
  p[b * 256 + c] = mx;
}

// ----------------------------------------------- head: tanh(p@Wp+bp), L2 norm
__global__ void __launch_bounds__(128)
head_kernel(const float* __restrict__ p, const float* __restrict__ Wp,
            const float* __restrict__ bp, float* __restrict__ out, int B) {
  __shared__ float t[64][128];
  __shared__ float nrm[64];
  const int jc = threadIdx.x;          // 128
  for (int b = 0; b < B; ++b) {
    float acc = bp[jc];
    for (int q = 0; q < 256; ++q) acc += p[b * 256 + q] * Wp[q * 128 + jc];
    t[b][jc] = tanhf(acc);
  }
  __syncthreads();
  if (jc < B) {
    float s = 0.f;
    for (int q = 0; q < 128; ++q) { const float v = t[jc][q]; s += v * v; }
    nrm[jc] = sqrtf(s) + 1e-9f;
  }
  __syncthreads();
  for (int b = 0; b < B; ++b) out[b * 128 + jc] = t[b][jc] / nrm[b];
}

// ---------------------------------------------------------------------------
extern "C" void kernel_launch(void* const* d_in, const int* in_sizes, int n_in,
                              void* d_out, int out_size, void* d_ws,
                              size_t ws_size, hipStream_t stream) {
  const float* x   = (const float*)d_in[0];
  const float* W1  = (const float*)d_in[2];
  const float* b1  = (const float*)d_in[3];
  const float* g1  = (const float*)d_in[4];
  const float* be1 = (const float*)d_in[5];
  const float* W2  = (const float*)d_in[6];
  const float* b2  = (const float*)d_in[7];
  const float* W3  = (const float*)d_in[8];
  const float* b3  = (const float*)d_in[9];
  const float* g3  = (const float*)d_in[10];
  const float* be3 = (const float*)d_in[11];
  const float* W4  = (const float*)d_in[12];
  const float* b4  = (const float*)d_in[13];
  const float* Wp  = (const float*)d_in[14];
  const float* bp  = (const float*)d_in[15];

  const int N  = in_sizes[0] / 3;      // 16384
  const int B  = in_sizes[1];          // 64
  const int NK = N * KNN;              // 163840
  const int pts = N / B;               // 256 (uniform segments per setup)

  // ---- carve workspace (256B aligned slabs) --------------------------------
  char* w = (char*)d_ws;
  auto carve = [&](size_t bytes) {
    char* q = w;
    w += (bytes + 255) & ~(size_t)255;
    return q;
  };
  int*      idx   = (int*)      carve((size_t)NK * 4);
  float*    hbuf  = (float*)    carve((size_t)NK * 64 * 4);
  float*    f1    = (float*)    carve((size_t)N * 64 * 4);
  float*    f2    = (float*)    carve((size_t)N * 64 * 4);
  float*    f3    = (float*)    carve((size_t)N * 64 * 4);
  float*    f4    = (float*)    carve((size_t)N * 64 * 4);
  _Float16* fh    = (_Float16*) carve((size_t)N * 64 * 2);
  float*    sq    = (float*)    carve((size_t)N * 4);
  _Float16* Wh    = (_Float16*) carve((size_t)128 * 64 * 2);
  float*    sums  = (float*)    carve(64 * 4);
  float*    sumsq = (float*)    carve(64 * 4);
  float*    scale = (float*)    carve(64 * 4);
  float*    shift = (float*)    carve(64 * 4);
  float*    pfeat = (float*)    carve((size_t)B * 256 * 4);
  (void)ws_size; (void)n_in; (void)out_size;

  const float invNK = 1.0f / (float)NK;

  // ---- layer 1: knn on xyz, edge MLP (6->64), BN, relu, max ----------------
  knn3_kernel<<<N / 256, 256, 0, stream>>>(x, idx, N);
  edge6_kernel<<<(NK + 255) / 256, 256, 0, stream>>>(x, idx, W1, b1, hbuf, NK);
  stats_kernel<<<64, 256, 0, stream>>>(hbuf, NK, sums, sumsq);
  bnfin_kernel<<<1, 64, 0, stream>>>(sums, sumsq, g1, be1, invNK, scale, shift);
  relumax_kernel<<<N, 64, 0, stream>>>(hbuf, scale, shift, 1, f1);
  castnorm_kernel<<<(N + 255) / 256, 256, 0, stream>>>(f1, fh, sq, N);

  // ---- layer 2: WMMA knn + WMMA edge GEMM (128->64), no BN -----------------
  castw_kernel<<<(128 * 64 + 255) / 256, 256, 0, stream>>>(W2, Wh, 128 * 64);
  knn64_kernel<<<N / 128, 128, 0, stream>>>(fh, sq, idx, N);
  edge128_kernel<<<NK / 64, 128, 0, stream>>>(fh, idx, Wh, b2, hbuf);
  relumax_kernel<<<N, 64, 0, stream>>>(hbuf, scale, shift, 0, f2);
  castnorm_kernel<<<(N + 255) / 256, 256, 0, stream>>>(f2, fh, sq, N);

  // ---- layer 3: WMMA knn + WMMA edge GEMM, with BN -------------------------
  castw_kernel<<<(128 * 64 + 255) / 256, 256, 0, stream>>>(W3, Wh, 128 * 64);
  knn64_kernel<<<N / 128, 128, 0, stream>>>(fh, sq, idx, N);
  edge128_kernel<<<NK / 64, 128, 0, stream>>>(fh, idx, Wh, b3, hbuf);
  stats_kernel<<<64, 256, 0, stream>>>(hbuf, NK, sums, sumsq);
  bnfin_kernel<<<1, 64, 0, stream>>>(sums, sumsq, g3, be3, invNK, scale, shift);
  relumax_kernel<<<N, 64, 0, stream>>>(hbuf, scale, shift, 1, f3);
  castnorm_kernel<<<(N + 255) / 256, 256, 0, stream>>>(f3, fh, sq, N);

  // ---- layer 4: WMMA knn + WMMA edge GEMM, no BN ---------------------------
  castw_kernel<<<(128 * 64 + 255) / 256, 256, 0, stream>>>(W4, Wh, 128 * 64);
  knn64_kernel<<<N / 128, 128, 0, stream>>>(fh, sq, idx, N);
  edge128_kernel<<<NK / 64, 128, 0, stream>>>(fh, idx, Wh, b4, hbuf);
  relumax_kernel<<<N, 64, 0, stream>>>(hbuf, scale, shift, 0, f4);

  // ---- aggregate + head ----------------------------------------------------
  segmax_kernel<<<B, 256, 0, stream>>>(f1, f2, f3, f4, pfeat, pts);
  head_kernel<<<1, 128, 0, stream>>>(pfeat, Wp, bp, (float*)d_out, B);
}
